// Voxel2Mesh_32040456028572
// MI455X (gfx1250) — matrix operand order repaired and checked by reading, not compiled, not run on hardware
//
#include <hip/hip_runtime.h>

typedef __attribute__((ext_vector_type(16))) _Float16 v16h;
typedef __attribute__((ext_vector_type(8)))  _Float16 v8h;
typedef __attribute__((ext_vector_type(8)))  float    v8f;
typedef unsigned int u32x4 __attribute__((ext_vector_type(4)));
typedef int          i32x8 __attribute__((ext_vector_type(8)));
typedef int          i32x4 __attribute__((ext_vector_type(4)));

#ifndef HAVE_TDM
#if defined(__has_builtin)
#if __has_builtin(__builtin_amdgcn_tensor_load_to_lds)
#define HAVE_TDM 1
#endif
#endif
#endif
#ifndef HAVE_TDM
#define HAVE_TDM 0
#endif

static inline int pad32i(int k) { return (k + 31) & ~31; }
static inline int cdiv(long a, long b) { return (int)((a + b - 1) / b); }

// ---------------------------------------------------------------------------
// Elementwise helpers
// ---------------------------------------------------------------------------
__global__ void k_f32_to_f16(const float* __restrict__ s, _Float16* __restrict__ d, int n) {
  int i = blockIdx.x * blockDim.x + threadIdx.x;
  if (i < n) d[i] = (_Float16)s[i];
}
__global__ void k_copy_f16(const _Float16* __restrict__ s, _Float16* __restrict__ d, int n) {
  int i = blockIdx.x * blockDim.x + threadIdx.x;
  if (i < n) d[i] = s[i];
}
__global__ void k_copy_f32(const float* __restrict__ s, float* __restrict__ d, int n) {
  int i = blockIdx.x * blockDim.x + threadIdx.x;
  if (i < n) d[i] = s[i];
}
__global__ void k_zero_f32(float* __restrict__ d, long n) {
  long i = blockIdx.x * (long)blockDim.x + threadIdx.x;
  if (i < n) d[i] = 0.f;
}
__global__ void k_add_f32(float* __restrict__ d, const float* __restrict__ s, int n) {
  int i = blockIdx.x * blockDim.x + threadIdx.x;
  if (i < n) d[i] += s[i];
}

// ---------------------------------------------------------------------------
// Weight packing (f32 -> f16, zero-padded K)
// ---------------------------------------------------------------------------
__global__ void k_pack_rm(const float* __restrict__ src, int N, int K, int Kpad,
                          _Float16* __restrict__ Bt) {
  long idx = blockIdx.x * (long)blockDim.x + threadIdx.x;
  if (idx >= (long)N * Kpad) return;
  int nn = (int)(idx / Kpad), k = (int)(idx % Kpad);
  Bt[idx] = (k < K) ? (_Float16)src[(long)nn * K + k] : (_Float16)0.f;
}
// LNS: einsum 'njc,ocj->no'; A column index k = j*C + c  =>  Bt[o][k] = w[o][c][j]
__global__ void k_pack_lns(const float* __restrict__ w, int O, int C, int Kpad,
                           _Float16* __restrict__ Bt) {
  long idx = blockIdx.x * (long)blockDim.x + threadIdx.x;
  if (idx >= (long)O * Kpad) return;
  int o = (int)(idx / Kpad), k = (int)(idx % Kpad);
  _Float16 v = (_Float16)0.f;
  if (k < 27 * C) { int j = k / C, c = k - j * C; v = (_Float16)w[((long)o * C + c) * 27 + j]; }
  Bt[idx] = v;
}
// gconv: A = [f | neigh], Bt[co][k] = w0[k][co] (k<Ci) or w1[k-Ci][co]
__global__ void k_pack_gconv(const float* __restrict__ w0, const float* __restrict__ w1,
                             int Ci, int Co, int Kpad, _Float16* __restrict__ Bt) {
  long idx = blockIdx.x * (long)blockDim.x + threadIdx.x;
  if (idx >= (long)Co * Kpad) return;
  int co = (int)(idx / Kpad), k = (int)(idx % Kpad);
  _Float16 v = (_Float16)0.f;
  if (k < Ci)            v = (_Float16)w0[(long)k * Co + co];
  else if (k < 2 * Ci)   v = (_Float16)w1[(long)(k - Ci) * Co + co];
  Bt[idx] = v;
}

// ---------------------------------------------------------------------------
// WMMA fragment assembly from LDS (16-bit A layout: two 8-half blocks / lane)
// ---------------------------------------------------------------------------
__device__ __forceinline__ v16h frag_A(const _Float16* Al, int wave, int lane) {
  const _Float16* ap = &Al[(wave * 16 + (lane & 15)) * 32 + ((lane >> 4) << 3)];
  v8h a0 = *(const v8h*)ap;          // K = 0..7   (lanes 0-15) / 8..15  (lanes 16-31)
  v8h a1 = *(const v8h*)(ap + 16);   // K = 16..23 (lanes 0-15) / 24..31 (lanes 16-31)
  return __builtin_shufflevector(a0, a1, 0, 1, 2, 3, 4, 5, 6, 7, 8, 9, 10, 11, 12, 13, 14, 15);
}
__device__ __forceinline__ v16h frag_B(const _Float16* Bl, int lane) {
  // column N = lane&15, K = 8h..8h+7 and 16+8h..16+8h+7 per half-wave (matches A)
  const _Float16* bp = &Bl[(lane & 15) * 32 + ((lane >> 4) << 3)];
  v8h b0 = *(const v8h*)bp;
  v8h b1 = *(const v8h*)(bp + 16);
  return __builtin_shufflevector(b0, b1, 0, 1, 2, 3, 4, 5, 6, 7, 8, 9, 10, 11, 12, 13, 14, 15);
}

// ---------------------------------------------------------------------------
// Implicit-GEMM conv3d (3x3x3, SAME), f16 NCDHW in -> f16 NCDHW out.
// block = 128 threads (4 wave32 waves), each wave owns one 16x16 C tile.
// ---------------------------------------------------------------------------
template <bool RELU>
__global__ void k_conv3d(const _Float16* __restrict__ in, const _Float16* __restrict__ wt,
                         const float* __restrict__ bias, _Float16* __restrict__ out,
                         int S, int Cin, int Cout, int Kpad) {
  __shared__ __align__(32) _Float16 Al[64 * 32];
  __shared__ __align__(32) _Float16 Bl[16 * 32];
  const int tid = threadIdx.x, wave = tid >> 5, lane = tid & 31;
  const int m0 = blockIdx.x * 64, n0 = blockIdx.y * 16;
  const int S2 = S * S, S3 = S2 * S, K = Cin * 27;
  v8f acc = {};
  for (int kb = 0; kb < Kpad; kb += 32) {
    for (int e = tid; e < 64 * 32; e += 128) {     // fused im2col gather -> LDS
      int r = e >> 5, kc = e & 31;
      int k = kb + kc, m = m0 + r;
      _Float16 v = (_Float16)0.f;
      if (k < K && m < S3) {
        int ci = k / 27, j = k - ci * 27;
        int dz = j / 9, rem = j - dz * 9;
        int dy = rem / 3, dx = rem - dy * 3;
        int z = m / S2, t2 = m - z * S2;
        int y = t2 / S, x = t2 - y * S;
        int zz = z + dz - 1, yy = y + dy - 1, xx = x + dx - 1;
        if ((unsigned)zz < (unsigned)S && (unsigned)yy < (unsigned)S && (unsigned)xx < (unsigned)S)
          v = in[(size_t)ci * S3 + (size_t)zz * S2 + (size_t)yy * S + xx];
      }
      Al[e] = v;
    }
    if (tid < 64) {                                 // vectorized weight-tile load (16x32)
      int r = tid >> 2, col8 = (tid & 3) << 3;
      int n = n0 + r;
      v8h v = {};
      if (n < Cout) v = *(const v8h*)&wt[(size_t)n * Kpad + kb + col8];
      *(v8h*)&Bl[tid << 3] = v;
    }
    if (kb + 32 < Kpad) __builtin_prefetch(&wt[(size_t)n0 * Kpad + kb + 32], 0, 1);
    __syncthreads();
    v16h a = frag_A(Al, wave, lane);
    v16h b = frag_B(Bl, lane);
    acc = __builtin_amdgcn_wmma_f32_16x16x32_f16(false, a, false, b, (short)0, acc, false, false);
    __syncthreads();
  }
  int n = n0 + (lane & 15);
  if (n < Cout) {
    float bv = bias[n];
    for (int r = 0; r < 8; ++r) {
      int m = m0 + wave * 16 + ((lane >> 4) << 3) + r;
      if (m < S3) {
        float v = acc[r] + bv;
        if (RELU) v = fmaxf(v, 0.f);
        out[(size_t)n * S3 + m] = (_Float16)v;
      }
    }
  }
}

// ---------------------------------------------------------------------------
// Generic WMMA GEMM: C[M,N] (f32 row-major) = A[M,Kpad] * Bt[N,Kpad]^T + bias
// 64x32 block tile; A staged via vector loads, B staged via the Tensor Data
// Mover (TENSOR_LOAD_TO_LDS, wave 0 issues, TENSORcnt waited) when available.
// ---------------------------------------------------------------------------
template <bool RELU>
__global__ void k_gemm(const _Float16* __restrict__ A, const _Float16* __restrict__ Bt,
                       const float* __restrict__ bias, float* __restrict__ outp,
                       int M, int N, int Kpad) {
  __shared__ __align__(32) _Float16 Al[64 * 32];
  __shared__ __align__(32) _Float16 Bl[32 * 32];
  const int tid = threadIdx.x, wave = tid >> 5, lane = tid & 31;
  const int m0 = blockIdx.x * 64, n0 = blockIdx.y * 32;
  v8f acc0 = {}, acc1 = {};
  for (int kb = 0; kb < Kpad; kb += 32) {
    // A tile: 64x32 halfs as 16-byte chunks, 2 per thread
#pragma unroll
    for (int j = 0; j < 2; ++j) {
      int c = tid + (j << 7);
      int r = c >> 2, col8 = (c & 3) << 3;
      int m = m0 + r;
      v8h v = {};
      if (m < M) v = *(const v8h*)&A[(size_t)m * Kpad + kb + col8];
      *(v8h*)&Al[c << 3] = v;
    }
#if HAVE_TDM
    // B tile: TDM DMA of a 2-D tile (32 rows x 32 halfs, row stride Kpad elems).
    // tensor_dim1 = N - n0 so out-of-range rows zero-fill the N tail.
    if (wave == 0) {
      unsigned long long ga = (unsigned long long)(uintptr_t)&Bt[(size_t)n0 * Kpad + kb];
      unsigned bl = (unsigned)(uintptr_t)&Bl[0];
      unsigned T0 = (unsigned)Kpad;
      unsigned T1 = (unsigned)(N - n0);
      u32x4 g0;
      g0[0] = 1u;                                            // count=1, user descriptor
      g0[1] = bl;                                            // lds_addr
      g0[2] = (unsigned)ga;                                  // global_addr[31:0]
      g0[3] = (unsigned)((ga >> 32) & 0x01ffffffu) | (2u << 30);  // addr[56:32] | type=2
      i32x8 g1;
      g1[0] = 0x00010000;                                    // data_size = 2 bytes
      g1[1] = (int)((T0 & 0xffffu) << 16);                   // tensor_dim0[15:0] @ bit48
      g1[2] = (int)(((T0 >> 16) & 0xffffu) | ((T1 & 0xffffu) << 16));   // dim0 hi | dim1 lo
      g1[3] = (int)(((T1 >> 16) & 0xffffu) | (32u << 16));   // dim1 hi | tile_dim0 = 32
      g1[4] = 32;                                            // tile_dim1 = 32 rows
      g1[5] = (int)T0;                                       // tensor_dim0_stride = Kpad
      g1[6] = 0;
      g1[7] = 0;
      i32x4 z4 = {0, 0, 0, 0};
#if __clang_major__ >= 23
      i32x8 z8 = {0, 0, 0, 0, 0, 0, 0, 0};
      __builtin_amdgcn_tensor_load_to_lds(g0, g1, z4, z4, z8, 0);
#else
      __builtin_amdgcn_tensor_load_to_lds(g0, g1, z4, z4, 0);
#endif
      __builtin_amdgcn_s_wait_tensorcnt(0);
    }
#else
    {
      int c = tid;                                           // 128 chunks = 32x32 halfs
      int r = c >> 2, col8 = (c & 3) << 3;
      int n = n0 + r;
      v8h v = {};
      if (n < N) v = *(const v8h*)&Bt[(size_t)n * Kpad + kb + col8];
      *(v8h*)&Bl[c << 3] = v;
    }
#endif
    if (kb + 32 < Kpad) __builtin_prefetch(&A[(size_t)m0 * Kpad + kb + 32], 0, 1);
    __syncthreads();
    v16h a = frag_A(Al, wave, lane);
    v16h b0 = frag_B(Bl, lane);
    v16h b1 = frag_B(Bl + 16 * 32, lane);
    acc0 = __builtin_amdgcn_wmma_f32_16x16x32_f16(false, a, false, b0, (short)0, acc0, false, false);
    acc1 = __builtin_amdgcn_wmma_f32_16x16x32_f16(false, a, false, b1, (short)0, acc1, false, false);
    __syncthreads();
  }
  int nA = n0 + (lane & 15), nB = nA + 16;
  for (int r = 0; r < 8; ++r) {
    int m = m0 + wave * 16 + ((lane >> 4) << 3) + r;
    if (m < M) {
      if (nA < N) {
        float v = acc0[r] + (bias ? bias[nA] : 0.f);
        if (RELU) v = fmaxf(v, 0.f);
        outp[(size_t)m * N + nA] = v;
      }
      if (nB < N) {
        float v = acc1[r] + (bias ? bias[nB] : 0.f);
        if (RELU) v = fmaxf(v, 0.f);
        outp[(size_t)m * N + nB] = v;
      }
    }
  }
}

// ---------------------------------------------------------------------------
// Pool / upconv / final conv
// ---------------------------------------------------------------------------
__global__ void k_maxpool(const _Float16* __restrict__ in, _Float16* __restrict__ out,
                          int C, int Si) {
  int So = Si >> 1, S3o = So * So * So;
  int idx = blockIdx.x * blockDim.x + threadIdx.x;
  if (idx >= C * S3o) return;
  int c = idx / S3o, m = idx - c * S3o;
  int z = m / (So * So), r = m - z * So * So;
  int y = r / So, x = r - y * So;
  const _Float16* p = in + (size_t)c * Si * Si * Si;
  float mx = -3.4e38f;
  for (int dz = 0; dz < 2; ++dz)
    for (int dy = 0; dy < 2; ++dy)
      for (int dx = 0; dx < 2; ++dx) {
        float v = (float)p[((size_t)(2 * z + dz) * Si + (2 * y + dy)) * Si + (2 * x + dx)];
        mx = fmaxf(mx, v);
      }
  out[idx] = (_Float16)mx;
}
// conv_transpose stride-2 kernel-2 "VALID": each output voxel maps to exactly one input voxel
__global__ void k_upconv(const _Float16* __restrict__ in, const float* __restrict__ w,
                         const float* __restrict__ b, _Float16* __restrict__ cat,
                         int Si, int Ci, int Co, int co_off) {
  int So = 2 * Si, S3o = So * So * So, S3i = Si * Si * Si;
  int idx = blockIdx.x * blockDim.x + threadIdx.x;
  if (idx >= Co * S3o) return;
  int co = idx / S3o, m = idx - co * S3o;
  int z = m / (So * So), r = m - z * So * So;
  int y = r / So, x = r - y * So;
  int zi = z >> 1, yi = y >> 1, xi = x >> 1;
  int woff = (z & 1) * 4 + (y & 1) * 2 + (x & 1);
  float acc = b[co];
  for (int ci = 0; ci < Ci; ++ci)
    acc += (float)in[(size_t)ci * S3i + ((size_t)zi * Si + yi) * Si + xi] *
           w[((size_t)co * Ci + ci) * 8 + woff];
  cat[(size_t)(co_off + co) * S3o + m] = (_Float16)acc;
}
__global__ void k_final_conv(const _Float16* __restrict__ xc, const float* __restrict__ w,
                             const float* __restrict__ b, float* __restrict__ out,
                             int Ci, int S) {
  int S3 = S * S * S;
  int idx = blockIdx.x * blockDim.x + threadIdx.x;
  if (idx >= 2 * S3) return;
  int c = idx / S3, m = idx - c * S3;
  float acc = b[c];
  for (int ci = 0; ci < Ci; ++ci) acc += (float)xc[(size_t)ci * S3 + m] * w[c * Ci + ci];
  out[idx] = acc;
}

// ---------------------------------------------------------------------------
// Icosphere topology (device-generated), graph ops
// ---------------------------------------------------------------------------
__global__ void k_init_ico(float* verts, int* faces) {
  if (threadIdx.x || blockIdx.x) return;
  const float t = 1.61803398875f;
  const float V[36] = {-1, t, 0, 1, t, 0, -1, -t, 0, 1, -t, 0, 0, -1, t, 0, 1, t,
                       0, -1, -t, 0, 1, -t, t, 0, -1, t, 0, 1, -t, 0, -1, -t, 0, 1};
  const int Fc[60] = {0, 11, 5, 0, 5, 1, 0, 1, 7, 0, 7, 10, 0, 10, 11, 1, 5, 9, 5, 11,
                      4, 11, 10, 2, 10, 7, 6, 7, 1, 8, 3, 9, 4, 3, 4, 2, 3, 2, 6, 3,
                      6, 8, 3, 8, 9, 4, 9, 5, 2, 4, 11, 6, 2, 10, 8, 6, 7, 9, 8, 1};
  for (int i = 0; i < 12; ++i) {
    float x = V[3 * i], y = V[3 * i + 1], z = V[3 * i + 2];
    float inv = rsqrtf(x * x + y * y + z * z);
    verts[3 * i] = x * inv; verts[3 * i + 1] = y * inv; verts[3 * i + 2] = z * inv;
  }
  for (int i = 0; i < 60; ++i) faces[i] = Fc[i];
}
__global__ void k_subdiv(const int* __restrict__ fin, int F, int nvb,
                         int* __restrict__ fout, int* __restrict__ ep) {
  int f = blockIdx.x * blockDim.x + threadIdx.x;
  if (f >= F) return;
  int a = fin[3 * f], b = fin[3 * f + 1], c = fin[3 * f + 2];
  int m = nvb + 3 * f;
  ep[2 * (3 * f + 0) + 0] = a; ep[2 * (3 * f + 0) + 1] = b;
  ep[2 * (3 * f + 1) + 0] = b; ep[2 * (3 * f + 1) + 1] = c;
  ep[2 * (3 * f + 2) + 0] = c; ep[2 * (3 * f + 2) + 1] = a;
  int* o = fout + 12 * f;
  o[0] = a; o[1] = m;     o[2] = m + 2;
  o[3] = b; o[4] = m + 1; o[5] = m;
  o[6] = c; o[7] = m + 2; o[8] = m + 1;
  o[9] = m; o[10] = m + 1; o[11] = m + 2;
}
__global__ void k_mid(const int* __restrict__ ep, int cnt, int nvb,
                      float* __restrict__ f, int C, int normalize) {
  int e = blockIdx.x * blockDim.x + threadIdx.x;
  if (e >= cnt) return;
  int a = ep[2 * e], b = ep[2 * e + 1], r = nvb + e;
  if (normalize) {
    float x = 0.5f * (f[a * 3 + 0] + f[b * 3 + 0]);
    float y = 0.5f * (f[a * 3 + 1] + f[b * 3 + 1]);
    float z = 0.5f * (f[a * 3 + 2] + f[b * 3 + 2]);
    float inv = rsqrtf(x * x + y * y + z * z + 1e-20f);
    f[r * 3 + 0] = x * inv; f[r * 3 + 1] = y * inv; f[r * 3 + 2] = z * inv;
  } else {
    for (int c = 0; c < C; ++c) f[(long)r * C + c] = 0.5f * (f[(long)a * C + c] + f[(long)b * C + c]);
  }
}
__global__ void k_degk(const int* __restrict__ faces, int F, float* __restrict__ deg) {
  int f = blockIdx.x * blockDim.x + threadIdx.x;
  if (f >= F) return;
  atomicAdd(&deg[faces[3 * f + 0]], 2.f);
  atomicAdd(&deg[faces[3 * f + 1]], 2.f);
  atomicAdd(&deg[faces[3 * f + 2]], 2.f);
}
__global__ void k_scatter(const int* __restrict__ faces, int F, const float* __restrict__ f,
                          int C, float* __restrict__ neigh) {
  long idx = blockIdx.x * (long)blockDim.x + threadIdx.x;
  if (idx >= (long)F * C) return;
  int fi = (int)(idx / C), c = (int)(idx % C);
  int a = faces[3 * fi], b = faces[3 * fi + 1], d = faces[3 * fi + 2];
  float va = f[(long)a * C + c], vb = f[(long)b * C + c], vd = f[(long)d * C + c];
  atomicAdd(&neigh[(long)a * C + c], vb + vd);
  atomicAdd(&neigh[(long)b * C + c], va + vd);
  atomicAdd(&neigh[(long)d * C + c], va + vb);
}
__global__ void k_build_gA(const float* __restrict__ f, const float* __restrict__ neigh,
                           const float* __restrict__ deg, int n, int C, int Kpad,
                           _Float16* __restrict__ A) {
  long idx = blockIdx.x * (long)blockDim.x + threadIdx.x;
  if (idx >= (long)n * Kpad) return;
  int m = (int)(idx / Kpad), k = (int)(idx % Kpad);
  float v = 0.f;
  if (k < C) v = f[(long)m * C + k];
  else if (k < 2 * C) v = neigh[(long)m * C + (k - C)] / fmaxf(deg[m], 1.f);
  A[idx] = (_Float16)v;
}
__global__ void k_concat3(const float* __restrict__ a, int Ca, const float* __restrict__ b,
                          int Cb, const float* __restrict__ c, int Cc, int n,
                          float* __restrict__ out) {
  int Ct = Ca + Cb + Cc;
  long idx = blockIdx.x * (long)blockDim.x + threadIdx.x;
  if (idx >= (long)n * Ct) return;
  int m = (int)(idx / Ct), k = (int)(idx % Ct);
  float v;
  if (k < Ca) v = a[(long)m * Ca + k];
  else if (k < Ca + Cb) v = b[(long)m * Cb + (k - Ca)];
  else v = c[(long)m * Cc + (k - Ca - Cb)];
  out[idx] = v;
}
// Build LNS feature matrix A[n][27*C] (k = j*C + c), trilinear sample of f16 volume
__global__ void k_lns_feat(const _Float16* __restrict__ vox, int C, int S,
                           const float* __restrict__ verts, int n, int Kpad,
                           _Float16* __restrict__ A) {
  long idx = blockIdx.x * (long)blockDim.x + threadIdx.x;
  if (idx >= (long)n * Kpad) return;
  int v = (int)(idx / Kpad), k = (int)(idx % Kpad);
  _Float16 outv = (_Float16)0.f;
  if (k < 27 * C) {
    int j = k / C, c = k - j * C;
    float step = 2.f / (float)S;
    float px = verts[v * 3 + 0] + ((float)(j / 9) - 1.f) * step;
    float py = verts[v * 3 + 1] + ((float)((j / 3) % 3) - 1.f) * step;
    float pz = verts[v * 3 + 2] + ((float)(j % 3) - 1.f) * step;
    float sm1 = (float)(S - 1);
    float gx = fminf(fmaxf((px + 1.f) * 0.5f * sm1, 0.f), sm1);
    float gy = fminf(fmaxf((py + 1.f) * 0.5f * sm1, 0.f), sm1);
    float gz = fminf(fmaxf((pz + 1.f) * 0.5f * sm1, 0.f), sm1);
    int x0 = (int)floorf(gx); if (x0 > S - 2) x0 = S - 2; if (x0 < 0) x0 = 0;
    int y0 = (int)floorf(gy); if (y0 > S - 2) y0 = S - 2; if (y0 < 0) y0 = 0;
    int z0 = (int)floorf(gz); if (z0 > S - 2) z0 = S - 2; if (z0 < 0) z0 = 0;
    float fx = gx - x0, fy = gy - y0, fz = gz - z0;
    const _Float16* ch = vox + (size_t)c * S * S * S;
    int S2 = S * S;
    float v000 = (float)ch[(size_t)z0 * S2 + y0 * S + x0];
    float v001 = (float)ch[(size_t)z0 * S2 + y0 * S + x0 + 1];
    float v010 = (float)ch[(size_t)z0 * S2 + (y0 + 1) * S + x0];
    float v011 = (float)ch[(size_t)z0 * S2 + (y0 + 1) * S + x0 + 1];
    float v100 = (float)ch[(size_t)(z0 + 1) * S2 + y0 * S + x0];
    float v101 = (float)ch[(size_t)(z0 + 1) * S2 + y0 * S + x0 + 1];
    float v110 = (float)ch[(size_t)(z0 + 1) * S2 + (y0 + 1) * S + x0];
    float v111 = (float)ch[(size_t)(z0 + 1) * S2 + (y0 + 1) * S + x0 + 1];
    float val = v000 * (1 - fz) * (1 - fy) * (1 - fx) + v001 * (1 - fz) * (1 - fy) * fx +
                v010 * (1 - fz) * fy * (1 - fx) + v011 * (1 - fz) * fy * fx +
                v100 * fz * (1 - fy) * (1 - fx) + v101 * fz * (1 - fy) * fx +
                v110 * fz * fy * (1 - fx) + v111 * fz * fy * fx;
    outv = (_Float16)val;
  }
  A[idx] = outv;
}

// ---------------------------------------------------------------------------
// Host orchestration
// ---------------------------------------------------------------------------
extern "C" void kernel_launch(void* const* d_in, const int* in_sizes, int n_in,
                              void* d_out, int out_size, void* d_ws, size_t ws_size,
                              hipStream_t stream) {
  (void)in_sizes; (void)ws_size;
  // ---- parameter unpack (x first, then params leaves in sorted-key order) ----
  int di = 0;
  auto nextp = [&]() -> const float* {
    int j = di < n_in ? di : (n_in > 0 ? n_in - 1 : 0);
    ++di;
    return (const float*)d_in[j];
  };
  const float* X = nextp();
  const float *down_b1[5], *down_b2[5], *down_w1[5], *down_w2[5];
  for (int i = 0; i < 5; ++i) { down_b1[i] = nextp(); down_b2[i] = nextp(); down_w1[i] = nextp(); down_w2[i] = nextp(); }
  const float *f2f_b[5][5], *f2f_w0[5][5], *f2f_w1[5][5];
  for (int i = 0; i < 5; ++i)
    for (int l = 0; l < 5; ++l) { f2f_b[i][l] = nextp(); f2f_w0[i][l] = nextp(); f2f_w1[i][l] = nextp(); }
  const float *f2v_b[5][3], *f2v_w0[5][3], *f2v_w1[5][3];
  for (int i = 0; i < 5; ++i)
    for (int l = 0; l < 3; ++l) { f2v_b[i][l] = nextp(); f2v_w0[i][l] = nextp(); f2v_w1[i][l] = nextp(); }
  const float* final_b = nextp();
  const float* final_w = nextp();
  const float *skip_b[5], *skip_w[5];
  for (int i = 0; i < 5; ++i) { skip_b[i] = nextp(); skip_w[i] = nextp(); }
  const float *up_ctb[5], *up_ctw[5], *up_b1[5], *up_b2[5], *up_w1[5], *up_w2[5];
  for (int i = 1; i <= 4; ++i) {
    up_ctb[i] = nextp(); up_ctw[i] = nextp();
    up_b1[i] = nextp(); up_b2[i] = nextp(); up_w1[i] = nextp(); up_w2[i] = nextp();
  }

  // ---- workspace bump allocator ----
  char* wp = (char*)d_ws;
  auto alloc = [&](size_t bytes) -> void* {
    void* r = (void*)wp;
    wp += (bytes + 255) & ~(size_t)255;
    return r;
  };
  const int S3 = 64 * 64 * 64;
  _Float16* x16  = (_Float16*)alloc((size_t)S3 * 2);
  _Float16* pool = (_Float16*)alloc((size_t)16 * 32 * 32 * 32 * 2);
  _Float16* tmp  = (_Float16*)alloc((size_t)16 * S3 * 2);
  _Float16* enc[5];
  enc[0] = (_Float16*)alloc((size_t)16 * S3 * 2);
  enc[1] = (_Float16*)alloc((size_t)32 * 32768 * 2);
  enc[2] = (_Float16*)alloc((size_t)64 * 4096 * 2);
  enc[3] = (_Float16*)alloc((size_t)128 * 512 * 2);
  enc[4] = (_Float16*)alloc((size_t)256 * 64 * 2);
  _Float16* cat = (_Float16*)alloc((size_t)32 * S3 * 2);
  _Float16* dec[5] = {nullptr, nullptr, nullptr, nullptr, nullptr};
  dec[1] = (_Float16*)alloc((size_t)128 * 512 * 2);
  dec[2] = (_Float16*)alloc((size_t)64 * 4096 * 2);
  dec[3] = (_Float16*)alloc((size_t)32 * 32768 * 2);
  dec[4] = (_Float16*)alloc((size_t)16 * S3 * 2);
  _Float16* wpack = (_Float16*)alloc((size_t)256 * 6912 * 2);
  _Float16* bt    = (_Float16*)alloc((size_t)256 * 6912 * 2);
  const int NVMAX = 81912, FMAX = 81920;
  _Float16* Abuf  = (_Float16*)alloc((size_t)NVMAX * 448 * 2);  // max of LNS / gconv A
  int* facesA = (int*)alloc((size_t)FMAX * 3 * 4);
  int* facesB = (int*)alloc((size_t)FMAX * 3 * 4);
  int* ep     = (int*)alloc((size_t)61440 * 2 * 4);
  float* verts   = (float*)alloc((size_t)NVMAX * 3 * 4);
  float* latent  = (float*)alloc((size_t)NVMAX * 32 * 4);
  float* deg     = (float*)alloc((size_t)NVMAX * 4);
  float* fbuf    = (float*)alloc((size_t)NVMAX * 64 * 4);
  float* gout    = (float*)alloc((size_t)NVMAX * 32 * 4);
  float* neigh   = (float*)alloc((size_t)NVMAX * 64 * 4);
  float* skipped = (float*)alloc((size_t)NVMAX * 16 * 4 + (size_t)312 * 256 * 4);
  float* gv1     = (float*)alloc((size_t)NVMAX * 16 * 4);
  float* gv2     = (float*)alloc((size_t)NVMAX * 16 * 4);

  auto run_gemm = [&](const _Float16* A, const _Float16* B, const float* bias, float* C,
                      int M, int N, int Kp, bool relu) {
    dim3 g(cdiv(M, 64), cdiv(N, 32));
    if (relu) k_gemm<true><<<g, 128, 0, stream>>>(A, B, bias, C, M, N, Kp);
    else      k_gemm<false><<<g, 128, 0, stream>>>(A, B, bias, C, M, N, Kp);
  };
  auto run_conv = [&](const _Float16* inp, const float* w, const float* b, _Float16* outp,
                      int Ssz, int Cin, int Cout, bool relu) {
    int K = Cin * 27, Kp = pad32i(K);
    k_pack_rm<<<cdiv((long)Cout * Kp, 256), 256, 0, stream>>>(w, Cout, K, Kp, wpack);
    dim3 g(cdiv((long)Ssz * Ssz * Ssz, 64), cdiv(Cout, 16));
    if (relu) k_conv3d<true><<<g, 128, 0, stream>>>(inp, wpack, b, outp, Ssz, Cin, Cout, Kp);
    else      k_conv3d<false><<<g, 128, 0, stream>>>(inp, wpack, b, outp, Ssz, Cin, Cout, Kp);
  };

  // ---- U-Net encoder ----
  const int ch[6] = {1, 16, 32, 64, 128, 256};
  k_f32_to_f16<<<cdiv(S3, 256), 256, 0, stream>>>(X, x16, S3);
  const _Float16* cin = x16;
  int S = 64;
  for (int i = 0; i < 5; ++i) {
    if (i > 0) {
      int So = S >> 1;
      k_maxpool<<<cdiv((long)ch[i] * So * So * So, 256), 256, 0, stream>>>(enc[i - 1], pool, ch[i], S);
      S = So; cin = pool;
    }
    run_conv(cin, down_w1[i], down_b1[i], tmp, S, ch[i], ch[i + 1], true);
    run_conv(tmp, down_w2[i], down_b2[i], enc[i], S, ch[i + 1], ch[i + 1], true);
  }

  // ---- sphere init: ico 12/20, two normalized subdivisions -> 312 verts / 320 faces ----
  k_init_ico<<<1, 32, 0, stream>>>(verts, facesA);
  k_subdiv<<<1, 32, 0, stream>>>(facesA, 20, 12, facesB, ep);
  k_mid<<<1, 64, 0, stream>>>(ep, 60, 12, verts, 3, 1);
  k_subdiv<<<2, 64, 0, stream>>>(facesB, 80, 72, facesA, ep);
  k_mid<<<4, 64, 0, stream>>>(ep, 240, 72, verts, 3, 1);
  int* faces = facesA;
  int* facesO = facesB;
  int nv = 312, F = 320;
  bool hasLat = false;

  auto run_gconv = [&](const float* fin, int Ci, int Co, const float* w0, const float* w1,
                       const float* b, float* foutP, bool relu) {
    k_zero_f32<<<cdiv((long)nv * Ci, 256), 256, 0, stream>>>(neigh, (long)nv * Ci);
    k_scatter<<<cdiv((long)F * Ci, 256), 256, 0, stream>>>(faces, F, fin, Ci, neigh);
    int Kp2 = pad32i(2 * Ci);
    k_build_gA<<<cdiv((long)nv * Kp2, 256), 256, 0, stream>>>(fin, neigh, deg, nv, Ci, Kp2, Abuf);
    k_pack_gconv<<<cdiv((long)Co * Kp2, 256), 256, 0, stream>>>(w0, w1, Ci, Co, Kp2, bt);
    run_gemm(Abuf, bt, b, foutP, nv, Co, Kp2, relu);
  };

  // ---- decoder + graph refinement loop ----
  const _Float16* xc = enc[4];
  int Sx = 4, Cxc = 256;
  for (int i = 0; i <= 4; ++i) {
    if (i > 0) {
      int ciu = 256 >> (i - 1), cou = ciu >> 1;
      int Sin = 4 << (i - 1), So = Sin << 1;
      int S3o = So * So * So;
      k_copy_f16<<<cdiv((long)cou * S3o, 256), 256, 0, stream>>>(enc[4 - i], cat, cou * S3o);
      k_upconv<<<cdiv((long)cou * S3o, 256), 256, 0, stream>>>(xc, up_ctw[i], up_ctb[i], cat, Sin, ciu, cou, cou);
      run_conv(cat, up_w1[i], up_b1[i], tmp, So, ciu, cou, true);
      run_conv(tmp, up_w2[i], up_b2[i], dec[i], So, cou, cou, true);
      xc = dec[i]; Sx = So; Cxc = cou;
      // unpool: subdivide sphere, midpoint-interp verts and latent
      k_subdiv<<<cdiv(F, 64), 64, 0, stream>>>(faces, F, nv, facesO, ep);
      k_mid<<<cdiv(3 * F, 64), 64, 0, stream>>>(ep, 3 * F, nv, verts, 3, 0);
      if (hasLat) k_mid<<<cdiv(3 * F, 64), 64, 0, stream>>>(ep, 3 * F, nv, latent, 32, 0);
      { int* t = faces; faces = facesO; facesO = t; }
      nv += 3 * F; F *= 4;
    }
    // vertex degrees for this topology level
    k_zero_f32<<<cdiv(nv, 256), 256, 0, stream>>>(deg, nv);
    k_degk<<<cdiv(F, 256), 256, 0, stream>>>(faces, F, deg);
    // LNS skip: trilinear 27-neighborhood sample -> GEMM (einsum njc,ocj->no)
    int C = Cxc, K = 27 * C, Kp = pad32i(K);
    k_lns_feat<<<cdiv((long)nv * Kp, 256), 256, 0, stream>>>(xc, C, Sx, verts, nv, Kp, Abuf);
    k_pack_lns<<<cdiv((long)C * Kp, 256), 256, 0, stream>>>(skip_w[i], C, C, Kp, bt);
    run_gemm(Abuf, bt, skip_b[i], skipped, nv, C, Kp, false);
    // input features = [latent? | skipped | verts]
    int CL = hasLat ? 32 : 0;
    int Cin0 = CL + C + 3;
    k_concat3<<<cdiv((long)nv * Cin0, 256), 256, 0, stream>>>(hasLat ? latent : skipped, CL,
                                                              skipped, C, verts, 3, nv, fbuf);
    // f2f: 5 graph-conv layers (relu on all but last)
    float* fa = fbuf;
    float* fb = gout;
    int ci = Cin0;
    for (int l = 0; l < 5; ++l) {
      run_gconv(fa, ci, 32, f2f_w0[i][l], f2f_w1[i][l], f2f_b[i][l], fb, l < 4);
      float* t = fa; fa = fb; fb = t; ci = 32;
    }
    k_copy_f32<<<cdiv((long)nv * 32, 256), 256, 0, stream>>>(fa, latent, nv * 32);
    hasLat = true;
    // f2v: 32 -> 16 -> 8 -> 3
    run_gconv(latent, 32, 16, f2v_w0[i][0], f2v_w1[i][0], f2v_b[i][0], gv1, true);
    run_gconv(gv1, 16, 8, f2v_w0[i][1], f2v_w1[i][1], f2v_b[i][1], gv2, true);
    run_gconv(gv2, 8, 3, f2v_w0[i][2], f2v_w1[i][2], f2v_b[i][2], gv1, false);
    k_add_f32<<<cdiv((long)nv * 3, 256), 256, 0, stream>>>(verts, gv1, nv * 3);
  }

  // ---- outputs: [verts (flat) | voxel_pred 2x64^3] ----
  float* out = (float*)d_out;
  int voxoff = out_size - 2 * S3;
  if (voxoff < 0) voxoff = 0;
  if (out_size >= 2 * S3)
    k_final_conv<<<cdiv(2 * S3, 256), 256, 0, stream>>>(dec[4], final_w, final_b, out + voxoff, 16, 64);
  int nvc = nv * 3;
  if (nvc > voxoff) nvc = voxoff;
  if (nvc > 0) k_copy_f32<<<cdiv(nvc, 256), 256, 0, stream>>>(verts, out, nvc);
}
